// MultiHeadAttentionBlock_42314017800885
// MI455X (gfx1250) — matrix operand order repaired and checked by reading, not compiled, not run on hardware
//
#include <hip/hip_runtime.h>
#include <hip/hip_bf16.h>

// MHA block for MI455X (gfx1250, wave32, WMMA).
// All GEMMs run on v_wmma_f32_16x16x32_bf16 with f32 accumulation.
// Attention K/V tiles are staged into LDS with double-buffered
// GLOBAL_LOAD_ASYNC_TO_LDS_B128 (ASYNCcnt) shared by the whole workgroup.
// Softmax denominator is accumulated on the matrix pipe (P @ ones).
// Mask: packed to 1 bit/key once, then 1 gather + ds_bpermute per tile.
// Requires ws_size >= 65 MiB.

typedef __bf16 bf16;
typedef __attribute__((ext_vector_type(16))) __bf16 v16bf;
typedef __attribute__((ext_vector_type(8)))  __bf16 v8bf;
typedef __attribute__((ext_vector_type(4)))  __bf16 v4bf;
typedef __attribute__((ext_vector_type(8)))  float  v8f;

#define B_  2
#define S_  2048
#define D_  1024
#define H_  16
#define DK_ 64

// 1/sqrt(dk) * log2(e): softmax runs in the exp2 domain (v_exp_f32 is exp2).
#define QSCALE 0.18033688011112042f

// ---------- helpers ----------

__device__ __forceinline__ bf16 f2bf(float f) {
  union { float f; unsigned u; } x; x.f = f;
  unsigned r = x.u + 0x7FFFu + ((x.u >> 16) & 1u);   // round-to-nearest-even
  unsigned short h = (unsigned short)(r >> 16);
  union { unsigned short s; bf16 b; } y; y.s = h;
  return y.b;
}

// A-operand (16x32 MxK, 16-bit): lane L: m=L&15, half=L>>4.
// VGPR 0-3 hold K = half*8 + 0..7 ; VGPR 4-7 hold K = 16 + half*8 + 0..7.
__device__ __forceinline__ v16bf load_a_tile(const bf16* __restrict__ p, int ld, int lane) {
  const int m = lane & 15, hf = lane >> 4;
  const bf16* q = p + (size_t)m * ld + hf * 8;
  v8bf lo = *(const v8bf*)q;
  v8bf hi = *(const v8bf*)(q + 16);
  v16bf r;
#pragma unroll
  for (int i = 0; i < 8; ++i) { r[i] = lo[i]; r[i + 8] = hi[i]; }
  return r;
}

// B-operand (32x16 KxN, 16-bit): lane L: n=L&15, half=L>>4.
// VGPR 0-3 hold K = half*8 + 0..7 ; VGPR 4-7 hold K = 16 + half*8 + 0..7. Source "transposed": B[k][n] = T[n][k].
__device__ __forceinline__ v16bf load_bt_tile(const bf16* __restrict__ p, int ld, int lane) {
  const int n = lane & 15, hf = lane >> 4;
  const bf16* q = p + (size_t)n * ld + hf * 8;
  v8bf lo = *(const v8bf*)q;
  v8bf hi = *(const v8bf*)(q + 16);
  v16bf r;
#pragma unroll
  for (int i = 0; i < 8; ++i) { r[i] = lo[i]; r[i + 8] = hi[i]; }
  return r;
}

__device__ __forceinline__ v8f wmma_bf16(v16bf a, v16bf b, v8f c) {
  return __builtin_amdgcn_wmma_f32_16x16x32_bf16(false, a, false, b, (short)0, c,
                                                 false, false);
}

// async global -> LDS, 16 bytes per lane (tracked with ASYNCcnt)
__device__ __forceinline__ void async_load_b128(unsigned lds_off, const bf16* gbase,
                                                unsigned goff_bytes) {
  asm volatile("global_load_async_to_lds_b128 %0, %1, %2 offset:0"
               :: "v"(lds_off), "v"(goff_bytes), "s"(gbase)
               : "memory");
}

// ---------- fp32 -> bf16 conversion (vector x4) ----------

__global__ __launch_bounds__(256)
void cvt_bf16_kernel(const float* __restrict__ src, bf16* __restrict__ dst, int n4) {
  int i = blockIdx.x * blockDim.x + threadIdx.x;
  if (i >= n4) return;
  const float4 f = ((const float4*)src)[i];
  v4bf o;
  o[0] = f2bf(f.x); o[1] = f2bf(f.y); o[2] = f2bf(f.z); o[3] = f2bf(f.w);
  ((v4bf*)dst)[i] = o;
}

// ---------- mask [B,S,S] int32 -> 1 bit per key (32x less HBM, read once) ----

__global__ __launch_bounds__(256)
void mask_pack_kernel(const int* __restrict__ mask, unsigned* __restrict__ bits, int n) {
  const int i = blockIdx.x * blockDim.x + threadIdx.x;
  const int pred = (i < n) ? mask[i] : 0;
  const unsigned b = (unsigned)__ballot(pred != 0);   // wave32: low 32 bits
  if ((threadIdx.x & 31) == 0 && i < n) bits[i >> 5] = b;
}

// ---------- Y = X @ W^T  (M=4096, N=1024, K=1024) ----------
// 4 waves/block; each wave computes a 64x64 tile (16 WMMA per K-step).
// MODE 0: Q proj -> bf16 [B,H,S,dk], scaled by log2(e)/sqrt(dk)
// MODE 1: K proj -> bf16 [B,H,S,dk]
// MODE 2: V proj -> bf16 [B,H,dk,S] (transposed for PV B-operand)
// MODE 3: O proj -> f32  [B,S,D]  (d_out)

template <int MODE>
__global__ __launch_bounds__(128)
void gemm_bt_kernel(const bf16* __restrict__ A, const bf16* __restrict__ W,
                    void* __restrict__ out) {
  const int lane  = threadIdx.x & 31;
  const int wave  = threadIdx.x >> 5;
  const int mbase = blockIdx.y * 128 + (wave >> 1) * 64;
  const int nbase = blockIdx.x * 128 + (wave & 1) * 64;

  v8f c[4][4] = {};

  for (int kk = 0; kk < D_; kk += 32) {
    v16bf a[4];
#pragma unroll
    for (int mt = 0; mt < 4; ++mt)
      a[mt] = load_a_tile(A + (size_t)(mbase + mt * 16) * D_ + kk, D_, lane);
#pragma unroll
    for (int nt = 0; nt < 4; ++nt) {
      v16bf bt = load_bt_tile(W + (size_t)(nbase + nt * 16) * D_ + kk, D_, lane);
#pragma unroll
      for (int mt = 0; mt < 4; ++mt)
        c[mt][nt] = wmma_bf16(a[mt], bt, c[mt][nt]);
    }
  }

  const int nl = lane & 15, hf = lane >> 4;
#pragma unroll
  for (int mt = 0; mt < 4; ++mt) {
#pragma unroll
    for (int nt = 0; nt < 4; ++nt) {
#pragma unroll
      for (int r = 0; r < 8; ++r) {
        const int gm = mbase + mt * 16 + r + 8 * hf;  // global row in [0, B*S)
        const int b  = gm >> 11;                      // S_ = 2048
        const int s  = gm & (S_ - 1);
        const int n  = nbase + nt * 16 + nl;
        const float v = c[mt][nt][r];
        if (MODE == 0) {
          const int h = n >> 6, d = n & (DK_ - 1);
          ((bf16*)out)[(((size_t)(b * H_ + h)) * S_ + s) * DK_ + d] = f2bf(v * QSCALE);
        } else if (MODE == 1) {
          const int h = n >> 6, d = n & (DK_ - 1);
          ((bf16*)out)[(((size_t)(b * H_ + h)) * S_ + s) * DK_ + d] = f2bf(v);
        } else if (MODE == 2) {
          const int h = n >> 6, d = n & (DK_ - 1);
          ((bf16*)out)[(((size_t)(b * H_ + h)) * DK_ + d) * S_ + s] = f2bf(v);
        } else {
          ((float*)out)[(size_t)gm * D_ + n] = v;
        }
      }
    }
  }
}

// ---------- flash attention ----------
// 4 waves/block, one 32-row q strip per wave, all waves share one (b,h).
// K/V tiles (32 keys) double-buffered in LDS via async copies; softmax in
// exp2 domain; mask via one gather + ds_bpermute; denominator via P @ ones.

__global__ __launch_bounds__(128)
void attn_kernel(const bf16* __restrict__ Qh, const bf16* __restrict__ Kh,
                 const bf16* __restrict__ Vt, const unsigned* __restrict__ mbits,
                 bf16* __restrict__ Xb) {
  __shared__ __align__(16) bf16 Kbuf[2][32 * DK_];  // [key][d]   8 KiB
  __shared__ __align__(16) bf16 Vbuf[2][DK_ * 32];  // [d][key]   8 KiB
  __shared__ __align__(16) bf16 Pbuf[4][32 * 32];   // per-wave P 8 KiB

  const int lane = threadIdx.x & 31;
  const int wave = threadIdx.x >> 5;
  const int nl = lane & 15, hf = lane >> 4;
  const int bh = blockIdx.y;               // b*H + h
  const int b = bh >> 4, h = bh & (H_ - 1);
  const int qbase = (blockIdx.x * 4 + wave) * 32;

  const bf16* Qp = Qh + ((size_t)bh * S_ + qbase) * DK_;
  const bf16* Kp = Kh + (size_t)bh * S_ * DK_;
  const bf16* Vp = Vt + (size_t)bh * DK_ * S_;
  const unsigned* Mb = mbits + ((size_t)b * S_ + qbase) * (S_ / 32);

  const unsigned klds[2] = { (unsigned)(uintptr_t)&Kbuf[0][0],
                             (unsigned)(uintptr_t)&Kbuf[1][0] };
  const unsigned vlds[2] = { (unsigned)(uintptr_t)&Vbuf[0][0],
                             (unsigned)(uintptr_t)&Vbuf[1][0] };

  // per-wave quarter of one K/V tile pair: 4 async b128 instructions
  const int idx0 = wave * 32 + lane;  // 0..127
  auto stage_tile = [&](int kbn, int buf) {
#pragma unroll
    for (int rep = 0; rep < 2; ++rep) {            // K: 32x64, 16B chunks
      const int idx = idx0 + rep * 128;
      const int row = idx >> 3, c = idx & 7;
      async_load_b128(klds[buf] + (unsigned)((row * DK_ + c * 8) * 2),
                      Kp, (unsigned)(((kbn + row) * DK_ + c * 8) * 2));
    }
#pragma unroll
    for (int rep = 0; rep < 2; ++rep) {            // V: 64x32, 16B chunks
      const int idx = idx0 + rep * 128;
      const int d = idx >> 2, c = idx & 3;
      async_load_b128(vlds[buf] + (unsigned)((d * 32 + c * 8) * 2),
                      Vp, (unsigned)((d * S_ + kbn + c * 8) * 2));
    }
  };

  // Q strip 32 x 64 -> 2 m-tiles x 2 K-chunks of A operands
  v16bf qa[2][2];
#pragma unroll
  for (int mt = 0; mt < 2; ++mt)
#pragma unroll
    for (int kc = 0; kc < 2; ++kc)
      qa[mt][kc] = load_a_tile(Qp + (size_t)(mt * 16) * DK_ + kc * 32, DK_, lane);

  // all-ones B tile: every column of P @ ones is the row sum of P
  v16bf ones_b;
#pragma unroll
  for (int i = 0; i < 16; ++i) ones_b[i] = f2bf(1.0f);

  v8f o[2][4] = {};
  v8f osum[2] = {};          // running softmax denominator (lane-replicated)
  float rmax[2][8];
#pragma unroll
  for (int mt = 0; mt < 2; ++mt)
#pragma unroll
    for (int r = 0; r < 8; ++r) rmax[mt][r] = -__builtin_inff();

  stage_tile(0, 0);  // prologue: fill buffer 0

  for (int kb = 0; kb < S_; kb += 32) {
    const int cur = (kb >> 5) & 1;

    // fence current tile (own quarter), then whole block (all quarters +
    // everyone done reading the buffer we are about to overwrite)
    asm volatile("s_wait_asynccnt 0x0" ::: "memory");
    __syncthreads();
    if (kb + 32 < S_) stage_tile(kb + 32, cur ^ 1);  // overlap with compute

    // mask dwords for all 32 rows of this strip: lane L holds row L's bits
    // (single global_load_b32; latency hidden under the score WMMAs)
    const unsigned rowbits = Mb[(size_t)lane * (S_ / 32) + (kb >> 5)];

    // ---- scores: 2 m-tiles x 2 key-tiles from LDS K ----
    v16bf bk[2][2];
#pragma unroll
    for (int j = 0; j < 2; ++j)
#pragma unroll
      for (int kc = 0; kc < 2; ++kc)
        bk[j][kc] = load_bt_tile(&Kbuf[cur][(16 * j) * DK_ + kc * 32], DK_, lane);

    v8f sc[2][2];
#pragma unroll
    for (int mt = 0; mt < 2; ++mt)
#pragma unroll
      for (int j = 0; j < 2; ++j) {
        v8f z = {};
        z = wmma_bf16(qa[mt][0], bk[j][0], z);
        sc[mt][j] = wmma_bf16(qa[mt][1], bk[j][1], z);
      }

    // ---- mask: broadcast row dword via ds_bpermute, test 2 bits ----
#pragma unroll
    for (int mt = 0; mt < 2; ++mt)
#pragma unroll
      for (int r = 0; r < 8; ++r) {
        const unsigned bits =
            (unsigned)__shfl((int)rowbits, mt * 16 + r + 8 * hf);
        if (!((bits >> nl) & 1u))        sc[mt][0][r] = -1e9f;
        if (!((bits >> (16 + nl)) & 1u)) sc[mt][1][r] = -1e9f;
      }

    // ---- online softmax max in exp2 domain (row = 16-lane column group) ----
    float alpha[2][8];
#pragma unroll
    for (int mt = 0; mt < 2; ++mt)
#pragma unroll
      for (int r = 0; r < 8; ++r) {
        float mx = fmaxf(sc[mt][0][r], sc[mt][1][r]);
        mx = fmaxf(mx, __shfl_xor(mx, 1));
        mx = fmaxf(mx, __shfl_xor(mx, 2));
        mx = fmaxf(mx, __shfl_xor(mx, 4));
        mx = fmaxf(mx, __shfl_xor(mx, 8));
        const float mnew = fmaxf(rmax[mt][r], mx);
        alpha[mt][r] = exp2f(rmax[mt][r] - mnew);
        rmax[mt][r] = mnew;
        sc[mt][0][r] = exp2f(sc[mt][0][r] - mnew);
        sc[mt][1][r] = exp2f(sc[mt][1][r] - mnew);
      }

    // rescale running O and denominator
#pragma unroll
    for (int mt = 0; mt < 2; ++mt) {
#pragma unroll
      for (int nt = 0; nt < 4; ++nt)
#pragma unroll
        for (int r = 0; r < 8; ++r) o[mt][nt][r] *= alpha[mt][r];
#pragma unroll
      for (int r = 0; r < 8; ++r) osum[mt][r] *= alpha[mt][r];
    }

    // ---- P: C-layout -> row-major 32x32 bf16 in LDS -> A operands ----
#pragma unroll
    for (int mt = 0; mt < 2; ++mt)
#pragma unroll
      for (int r = 0; r < 8; ++r) {
        const int m = mt * 16 + r + 8 * hf;
        Pbuf[wave][m * 32 + nl]      = f2bf(sc[mt][0][r]);
        Pbuf[wave][m * 32 + 16 + nl] = f2bf(sc[mt][1][r]);
      }
    asm volatile("s_wait_dscnt 0" ::: "memory");
    v16bf pa[2];
#pragma unroll
    for (int mt = 0; mt < 2; ++mt)
      pa[mt] = load_a_tile(&Pbuf[wave][mt * 16 * 32], 32, lane);

    // ---- O += P @ V ; denominator += P @ ones (matrix pipe) ----
#pragma unroll
    for (int nt = 0; nt < 4; ++nt) {
      v16bf vb = load_bt_tile(&Vbuf[cur][(nt * 16) * 32], 32, lane);
#pragma unroll
      for (int mt = 0; mt < 2; ++mt)
        o[mt][nt] = wmma_bf16(pa[mt], vb, o[mt][nt]);
    }
#pragma unroll
    for (int mt = 0; mt < 2; ++mt)
      osum[mt] = wmma_bf16(pa[mt], ones_b, osum[mt]);
  }

  // ---- finalize: O / l (l already replicated across lanes), store bf16 ----
#pragma unroll
  for (int mt = 0; mt < 2; ++mt) {
    float inv[8];
#pragma unroll
    for (int r = 0; r < 8; ++r) inv[r] = 1.0f / osum[mt][r];
#pragma unroll
    for (int nt = 0; nt < 4; ++nt)
#pragma unroll
      for (int r = 0; r < 8; ++r) {
        const int m = mt * 16 + r + 8 * hf;
        Xb[((size_t)(b * S_ + qbase + m)) * D_ + h * DK_ + nt * 16 + nl] =
            f2bf(o[mt][nt][r] * inv[r]);
      }
  }
}

// ---------- host ----------

extern "C" void kernel_launch(void* const* d_in, const int* in_sizes, int n_in,
                              void* d_out, int out_size, void* d_ws, size_t ws_size,
                              hipStream_t stream) {
  (void)in_sizes; (void)n_in; (void)out_size; (void)ws_size;

  const float* q    = (const float*)d_in[0];
  const float* k    = (const float*)d_in[1];
  const float* v    = (const float*)d_in[2];
  const int*   mask = (const int*)d_in[3];
  const float* Wq   = (const float*)d_in[4];
  const float* Wk   = (const float*)d_in[5];
  const float* Wv   = (const float*)d_in[6];
  const float* Wo   = (const float*)d_in[7];

  const size_t NE = (size_t)B_ * S_ * D_;   // 4,194,304
  const size_t WE = (size_t)D_ * D_;        // 1,048,576
  const size_t ME = (size_t)B_ * S_ * S_;   // 8,388,608 mask elements

  bf16* qb  = (bf16*)d_ws;       // bf16 activations
  bf16* kb  = qb + NE;
  bf16* vb  = kb + NE;
  bf16* wqb = vb + NE;           // bf16 weights
  bf16* wkb = wqb + WE;
  bf16* wvb = wkb + WE;
  bf16* wob = wvb + WE;
  bf16* Qh  = wob + WE;          // [B,H,S,dk], pre-scaled by log2(e)/8
  bf16* Kh  = Qh + NE;           // [B,H,S,dk]
  bf16* Vt  = Kh + NE;           // [B,H,dk,S]
  bf16* Xb  = Vt + NE;           // [B,S,D] attention output
  unsigned* mbits = (unsigned*)(Xb + NE);   // packed mask bits, 1 MiB
  // total = 7*NE + 4*WE bf16 + ME/8 bytes = 65 MiB

  // 1) fp32 -> bf16, mask -> bits
  cvt_bf16_kernel<<<(int)(NE / 4 / 256), 256, 0, stream>>>(q, qb, (int)(NE / 4));
  cvt_bf16_kernel<<<(int)(NE / 4 / 256), 256, 0, stream>>>(k, kb, (int)(NE / 4));
  cvt_bf16_kernel<<<(int)(NE / 4 / 256), 256, 0, stream>>>(v, vb, (int)(NE / 4));
  cvt_bf16_kernel<<<(int)(WE / 4 / 256), 256, 0, stream>>>(Wq, wqb, (int)(WE / 4));
  cvt_bf16_kernel<<<(int)(WE / 4 / 256), 256, 0, stream>>>(Wk, wkb, (int)(WE / 4));
  cvt_bf16_kernel<<<(int)(WE / 4 / 256), 256, 0, stream>>>(Wv, wvb, (int)(WE / 4));
  cvt_bf16_kernel<<<(int)(WE / 4 / 256), 256, 0, stream>>>(Wo, wob, (int)(WE / 4));
  mask_pack_kernel<<<(int)(ME / 256), 256, 0, stream>>>(mask, mbits, (int)ME);

  // 2) projections (128x128 per block, 64x64 per wave)
  dim3 ggrid(D_ / 128, (B_ * S_) / 128);
  gemm_bt_kernel<0><<<ggrid, 128, 0, stream>>>(qb, wqb, Qh);
  gemm_bt_kernel<1><<<ggrid, 128, 0, stream>>>(kb, wkb, Kh);
  gemm_bt_kernel<2><<<ggrid, 128, 0, stream>>>(vb, wvb, Vt);

  // 3) flash attention (4 q-strips of 32 rows per block)
  dim3 agrid(S_ / 128, B_ * H_);
  attn_kernel<<<agrid, 128, 0, stream>>>(Qh, Kh, Vt, mbits, Xb);

  // 4) output projection -> f32 d_out
  gemm_bt_kernel<3><<<ggrid, 128, 0, stream>>>(Xb, wob, (float*)d_out);
}
